// MoveNetModel_3032246911580
// MI455X (gfx1250) — matrix operand order repaired and checked by reading, not compiled, not run on hardware
//
#include <hip/hip_runtime.h>

typedef __attribute__((ext_vector_type(16))) __bf16 v16bf;
typedef __attribute__((ext_vector_type(8)))  float  v8f;

#define CIN   48
#define CMID  24
#define NCH   96        // 4 heads * 24 mid channels
#define KTOT  432       // 9 shifts * 48 cin
#define KPAD  448       // 14 K-blocks of 32
#define NPOS  576       // 24*24
#define CPAD  50        // padded cin stride in LDS (25 dwords -> conflict-free)
#define ZSLOT (NPOS * CPAD)   // index (in shorts) of the 48-short zero pad

// d_ws byte offsets
#define WS_WB   0                       // bf16 Wb[96][448]
#define WS_A1   (NCH * KPAD * 2)        // 86016: f32 A1[96]  (g*rsqrt(v+eps))
#define WS_B1   (WS_A1 + NCH * 4)       // f32 B1[96]  ((b1-m)*A1 + beta)
#define WS_W2   (WS_B1 + NCH * 4)       // f32 W2[21][24]
#define WS_B2   (WS_W2 + 21 * 24 * 4)   // f32 B2[21]

__device__ __forceinline__ unsigned short f2bf(float f) {
    unsigned int u = __builtin_bit_cast(unsigned int, f);
    u += 0x7fffu + ((u >> 16) & 1u);        // round-to-nearest-even
    return (unsigned short)(u >> 16);
}

// ---------------- prep: pack conv1 weights (bf16, shift-major K, zero-padded) ----
__global__ void pack_w1(const float* cw1, const float* kw1, const float* rw1,
                        const float* ow1, unsigned short* Wb) {
    int i = blockIdx.x * blockDim.x + threadIdx.x;
    if (i >= NCH * KPAD) return;
    int row = i / KPAD, k = i % KPAD;
    float val = 0.f;
    if (k < KTOT) {
        int head = row / CMID, oc = row % CMID;
        int s = k / CIN, cin = k - s * CIN;
        int dy = s / 3, dx = s - dy * 3;
        const float* w1 = head == 0 ? cw1 : head == 1 ? kw1 : head == 2 ? rw1 : ow1;
        val = w1[((oc * CIN + cin) * 3 + dy) * 3 + dx];
    }
    Wb[i] = f2bf(val);
}

struct ScArgs {
    const float* b1[4]; const float* g[4]; const float* be[4];
    const float* m[4];  const float* v[4]; const float* w2[4]; const float* b2[4];
    float* A1; float* B1; float* W2; float* B2;
};

__device__ __forceinline__ int co_head(int co) { return co == 0 ? 0 : co < 5 ? 1 : co < 13 ? 2 : 3; }
__device__ __forceinline__ int head_base(int h) { return h == 0 ? 0 : h == 1 ? 1 : h == 2 ? 5 : 13; }

__global__ void pack_sc(ScArgs a) {
    int t = threadIdx.x;
    if (t < NCH) {
        int h = t / CMID, c = t % CMID;
        float inv = rsqrtf(a.v[h][c] + 1e-5f);
        float sc  = a.g[h][c] * inv;
        a.A1[t] = sc;
        a.B1[t] = (a.b1[h][c] - a.m[h][c]) * sc + a.be[h][c];
    }
    for (int i = t; i < 21 * 24; i += blockDim.x) {
        int co = i / 24, c = i % 24;
        int h = co_head(co), j = co - head_base(h);
        a.W2[i] = a.w2[h][j * 24 + c];
    }
    if (t < 21) {
        int h = co_head(t), j = t - head_base(h);
        a.B2[t] = a.b2[h][j];
    }
}

// ---------------- main: fused 4-head conv3x3(WMMA) + BN + ReLU + conv1x1 + sigmoid
union Frag { v16bf v; unsigned int u[8]; };

__global__ __launch_bounds__(384) void conv_heads(
    const float* __restrict__ feat, const unsigned short* __restrict__ Wb,
    const float* __restrict__ A1, const float* __restrict__ B1,
    const float* __restrict__ W2, const float* __restrict__ B2,
    float* __restrict__ out, int Bn)
{
    __shared__ unsigned short xT[NPOS * CPAD + 48];  // [pos][cin] + 96B zero pad

    const int b   = blockIdx.x;
    const int tid = threadIdx.x;

    // stage image -> LDS as bf16 (float4 global loads, coalesced)
    const float4* xb4 = (const float4*)(feat + (size_t)b * (CIN * NPOS));
    for (int i = tid; i < CIN * NPOS / 4; i += 384) {
        float4 f = xb4[i];
        int cin = i / (NPOS / 4);
        int p   = (i - cin * (NPOS / 4)) * 4;       // 4 consecutive positions
        int adr = p * CPAD + cin;
        xT[adr]            = f2bf(f.x);
        xT[adr + CPAD]     = f2bf(f.y);
        xT[adr + 2 * CPAD] = f2bf(f.z);
        xT[adr + 3 * CPAD] = f2bf(f.w);
    }
    if (tid < 48) xT[ZSLOT + tid] = 0;
    __syncthreads();

    const int wave = tid >> 5, lane = tid & 31;
    const int half = lane >> 4;            // which 16-lane half
    const int l16  = lane & 15;

    const size_t oK = (size_t)Bn * NPOS;                 // kp base (floats)
    const size_t oR = oK + (size_t)Bn * 4 * NPOS;        // reg base
    const size_t oO = oR + (size_t)Bn * 8 * NPOS;        // off base

#pragma unroll 1
    for (int t3 = 0; t3 < 3; ++t3) {
        // opaque zero offset: defeats LICM (no hoist->spill of weight loads)
        // while keeping pointer provenance -> global_load (not flat_load).
        int zoff = 0;
        asm volatile("" : "+s"(zoff));
        const unsigned short* WbL = Wb + zoff;
        const float* A1L = A1 + zoff; const float* B1L = B1 + zoff;
        const float* W2L = W2 + zoff; const float* B2L = B2 + zoff;

        const int nt = wave * 3 + t3;       // 12 waves * 3 = 36 N-tiles
        const int n0 = nt * 16;
        const int n  = n0 + l16;
        const int y  = n / 24, x = n - y * 24;

        // per-shift LDS byte bases; OOB (SAME padding) and K-tail -> zero pad
        int sb[10];
#pragma unroll
        for (int s = 0; s < 9; ++s) {
            int dy = s / 3, dx = s - (s / 3) * 3;
            int iy = y + dy - 1, ix = x + dx - 1;
            bool valid = ((unsigned)iy < 24u) & ((unsigned)ix < 24u);
            sb[s] = valid ? ((iy * 24 + ix) * CPAD) * 2 : ZSLOT * 2;
        }
        sb[9] = ZSLOT * 2;                  // K-group g=27 (k>=432) reads zeros

        // 6 accumulators live across the whole K loop (48 VGPRs)
        v8f acc[6];
#pragma unroll
        for (int m = 0; m < 6; ++m) acc[m] = (v8f){};

        // K outer / M inner: one B fragment live at a time (low pressure)
#pragma unroll
        for (int kb = 0; kb < 14; ++kb) {
            // 16-element K-groups never cross a shift (48 = 3*16):
            // g = 2*kb + half, s = g/3, cin = 16*(g%3) + 2v  (all compile-time)
            const int g0 = 2 * kb,     s0 = g0 / 3, c0 = 16 * (g0 % 3);
            const int g1 = 2 * kb + 1, s1 = g1 / 3, c1 = 16 * (g1 % 3);
            int baseB = half ? (sb[s1] + 2 * c1) : (sb[s0] + 2 * c0);
            Frag bf;
#pragma unroll
            for (int v = 0; v < 8; ++v)   // ds_load_b32 base + immediate 4*v
                bf.u[v] = *(const unsigned int*)((const char*)xT + (baseB + 4 * v));

            // ---- stream A fragments (weights, L2-resident) and accumulate ----
#pragma unroll
            for (int m = 0; m < 6; ++m) {
                const unsigned short* wrow = WbL + (size_t)(m * 16 + l16) * KPAD;
                Frag af;
#pragma unroll
                for (int v = 0; v < 8; ++v) {
                    int kA = kb * 32 + 2 * v + half * 16;
                    af.u[v] = *(const unsigned int*)&wrow[kA];
                }
                acc[m] = __builtin_amdgcn_wmma_f32_16x16x32_bf16(
                             false, af.v, false, bf.v, (short)0, acc[m], false, false);
            }
        }

        // ---- fused BN + ReLU + 1x1-conv (channel indices constant after unroll)
        float outc[21];
#pragma unroll
        for (int i = 0; i < 21; ++i) outc[i] = 0.f;

#pragma unroll
        for (int m = 0; m < 6; ++m) {
#pragma unroll
            for (int r = 0; r < 8; ++r) {
                const int chlo = m * 16 + r, chhi = chlo + 8;
                float aS = half ? A1L[chhi] : A1L[chlo];
                float bS = half ? B1L[chhi] : B1L[chlo];
                float h  = fmaxf(acc[m][r] * aS + bS, 0.f);
                // SWAPX16: and=0x1f, xor=0x10 -> 0x401f (lane l <-> l^16, same n)
                float hp = __builtin_bit_cast(float,
                              __builtin_amdgcn_ds_swizzle(__builtin_bit_cast(int, h), 0x401f));
                float hlo = half ? hp : h;
                float hhi = half ? h  : hp;
#pragma unroll
                for (int t = 0; t < 2; ++t) {
                    const int   ch = t ? chhi : chlo;
                    const float hv = t ? hhi  : hlo;
                    const int hd = ch / CMID, c = ch - hd * CMID;
                    const int base = head_base(hd);
                    const int cnt  = hd == 0 ? 1 : hd == 1 ? 4 : 8;
#pragma unroll
                    for (int j = 0; j < 8; ++j)
                        if (j < cnt) outc[base + j] += hv * W2L[(base + j) * 24 + c];
                }
            }
        }

        // bias, sigmoid (co 0..4), store (halves split the 21 channels)
#pragma unroll
        for (int co = 0; co < 21; ++co) outc[co] += B2L[co];
#pragma unroll
        for (int co = 0; co < 5; ++co) outc[co] = 1.f / (1.f + expf(-outc[co]));

        if (!half) {
            out[(size_t)b * NPOS + n] = outc[0];                                   // center
#pragma unroll
            for (int co = 1; co < 5; ++co)                                         // kp
                out[oK + ((size_t)b * 4 + (co - 1)) * NPOS + n] = outc[co];
#pragma unroll
            for (int co = 5; co < 11; ++co)                                        // reg 0..5
                out[oR + ((size_t)b * 8 + (co - 5)) * NPOS + n] = outc[co];
        } else {
#pragma unroll
            for (int co = 11; co < 13; ++co)                                       // reg 6..7
                out[oR + ((size_t)b * 8 + (co - 5)) * NPOS + n] = outc[co];
#pragma unroll
            for (int co = 13; co < 21; ++co)                                       // off
                out[oO + ((size_t)b * 8 + (co - 13)) * NPOS + n] = outc[co];
        }
    }
}

// ---------------- argmax over kp heatmaps + offset gather -> norm / pix ----------
__global__ __launch_bounds__(256) void keypoints(const float* src, float* out, int Bn) {
    int wv   = (blockIdx.x * blockDim.x + threadIdx.x) >> 5;
    int lane = threadIdx.x & 31;
    if (wv >= Bn * 4) return;
    int b = wv >> 2, kpt = wv & 3;

    size_t oK = (size_t)Bn * NPOS;
    size_t oO = oK + (size_t)Bn * 4 * NPOS + (size_t)Bn * 8 * NPOS;
    const float* hm = src + oK + ((size_t)b * 4 + kpt) * NPOS;

    float best = -1e30f; int bidx = 0x7fffffff;
    for (int i = lane; i < NPOS; i += 32) {
        float v = hm[i];
        if (v > best) { best = v; bidx = i; }       // strict > keeps first max
    }
#pragma unroll
    for (int off = 16; off >= 1; off >>= 1) {
        float ov = __shfl_xor(best, off, 32);
        int   oi = __shfl_xor(bidx, off, 32);
        if (ov > best || (ov == best && oi < bidx)) { best = ov; bidx = oi; }
    }
    if (lane == 0) {
        int my = bidx / 24, mx = bidx - my * 24;
        const float* offp = src + oO + (size_t)b * 8 * NPOS;
        float ox = offp[(kpt * 2 + 0) * NPOS + bidx];
        float oy = offp[(kpt * 2 + 1) * NPOS + bidx];
        float fx = (mx + ox) * 4.f - 112.f;
        float fy = (my + oy) * 4.f;
        size_t oN = oO + (size_t)Bn * 8 * NPOS;     // norm base
        size_t oP = oN + (size_t)Bn * 8;            // pix base
        out[oN + (size_t)b * 8 + kpt * 2 + 0] = (fx + 112.f) / 336.f;
        out[oN + (size_t)b * 8 + kpt * 2 + 1] = fy / 224.f;
        out[oP + (size_t)b * 8 + kpt * 2 + 0] = fx;
        out[oP + (size_t)b * 8 + kpt * 2 + 1] = fy;
    }
}

extern "C" void kernel_launch(void* const* d_in, const int* in_sizes, int n_in,
                              void* d_out, int out_size, void* d_ws, size_t ws_size,
                              hipStream_t stream) {
    const float* feat = (const float*)d_in[0];
    const int Bn = in_sizes[0] / (CIN * NPOS);

    unsigned short* Wb = (unsigned short*)d_ws;
    float* A1 = (float*)((char*)d_ws + WS_A1);
    float* B1 = (float*)((char*)d_ws + WS_B1);
    float* W2 = (float*)((char*)d_ws + WS_W2);
    float* B2 = (float*)((char*)d_ws + WS_B2);

    // heads in dict order: c,k,r,o ; per head: w1,b1,g,be,m,v,w2,b2 at d_in[1+8h ..]
    ScArgs sa;
    const float* w1p[4];
    for (int h = 0; h < 4; ++h) {
        int base = 1 + h * 8;
        w1p[h]   = (const float*)d_in[base + 0];
        sa.b1[h] = (const float*)d_in[base + 1];
        sa.g[h]  = (const float*)d_in[base + 2];
        sa.be[h] = (const float*)d_in[base + 3];
        sa.m[h]  = (const float*)d_in[base + 4];
        sa.v[h]  = (const float*)d_in[base + 5];
        sa.w2[h] = (const float*)d_in[base + 6];
        sa.b2[h] = (const float*)d_in[base + 7];
    }
    sa.A1 = A1; sa.B1 = B1; sa.W2 = W2; sa.B2 = B2;

    pack_w1<<<(NCH * KPAD + 255) / 256, 256, 0, stream>>>(w1p[0], w1p[1], w1p[2], w1p[3], Wb);
    pack_sc<<<1, 256, 0, stream>>>(sa);
    conv_heads<<<Bn, 384, 0, stream>>>(feat, Wb, A1, B1, W2, B2, (float*)d_out, Bn);
    keypoints<<<(Bn * 4 * 32 + 255) / 256, 256, 0, stream>>>((const float*)d_out, (float*)d_out, Bn);
}
